// AutoCorrelation_21612275434097
// MI455X (gfx1250) — matrix operand order repaired and checked by reading, not compiled, not run on hardware
//
#include <hip/hip_runtime.h>

// ---------------------------------------------------------------------------
// AutoCorrelation (Autoformer) for MI455X / gfx1250, wave32 + WMMA bf16.
// B=8, L=4096, H=8, E=64  ->  D = H*E = 512 (contiguous innermost).
//
// mean_value[b,l] = (1/D) * sum_t <Q[b,t,:], K[b,(t-l)%L,:]>
//                 = (1/D) * sum over diagonal (t-s)%L of Gram(Q,K^T).
// Computed with v_wmma_f32_16x16x32_bf16 Gram tiles + LDS lag histogram.
// D is processed in two half-passes so the A-operand register array stays at
// 64 VGPRs (no scratch spills in the WMMA loop).
// ---------------------------------------------------------------------------

#define BB   8
#define LL   4096
#define DD   512          // H*E
#define TOPK 8            // int(log(4096)) = 8
#define LMASK 4095

typedef __attribute__((ext_vector_type(16))) __bf16 v16bf;
typedef __attribute__((ext_vector_type(8)))  float  v8f;

union ABTile {
    v16bf v;
    uint4 q[2];
};

__device__ __forceinline__ unsigned int f2bf(float f) {
    unsigned int u = __float_as_uint(f);
    // round-to-nearest-even truncation to bf16
    return (u + 0x7FFFu + ((u >> 16) & 1u)) >> 16;
}

// ---------------------------------------------------------------------------
// Kernel 1: pack Q,K -> bf16 workspace; zero mean_value accumulator.
// grid: (B*L*D)/4 threads.
// ---------------------------------------------------------------------------
__global__ void ac_prep(const float* __restrict__ q, const float* __restrict__ k,
                        unsigned short* __restrict__ qb, unsigned short* __restrict__ kb,
                        float* __restrict__ meanval) {
    unsigned int tid = blockIdx.x * 256u + threadIdx.x;
    if (tid < BB * LL) meanval[tid] = 0.0f;

    size_t i0 = (size_t)tid * 4u;
    float4 qq = *(const float4*)(q + i0);
    float4 kk = *(const float4*)(k + i0);
    uint2 pq, pk;
    pq.x = f2bf(qq.x) | (f2bf(qq.y) << 16);
    pq.y = f2bf(qq.z) | (f2bf(qq.w) << 16);
    pk.x = f2bf(kk.x) | (f2bf(kk.y) << 16);
    pk.y = f2bf(kk.z) | (f2bf(kk.w) << 16);
    *(uint2*)(qb + i0) = pq;
    *(uint2*)(kb + i0) = pk;
}

// ---------------------------------------------------------------------------
// Kernel 2: Gram tiles via WMMA bf16, diagonal-scatter into LDS lag bins.
// grid: B * (L/16) blocks of 256 threads (8 waves). Wave w handles s-tiles
// w, w+8, ... (32 tiles). D is covered in two passes of 8 k-chunks; the
// 8 A-chunks for the active pass live in a 64-VGPR register array reused
// across all 32 s-tiles.
// ---------------------------------------------------------------------------
__global__ __launch_bounds__(256)
void ac_corr_wmma(const unsigned short* __restrict__ qb,
                  const unsigned short* __restrict__ kb,
                  float* __restrict__ meanval) {
    __shared__ float lagAcc[LL];

    const int tid = threadIdx.x;
    for (int i = tid; i < LL; i += 256) lagAcc[i] = 0.0f;
    __syncthreads();

    const int b    = blockIdx.x >> 8;          // 256 t-tiles per batch
    const int t0   = (blockIdx.x & 255) << 4;
    const int lane = tid & 31;
    const int wave = tid >> 5;
    const int m    = lane & 15;                // A row / B column per lane
    const int hi   = lane >> 4;                // 0: lanes 0-15, 1: lanes 16-31
    const int ka   = hi ? 8 : 0;               // A-matrix K base (ISA layout)
    const int kbb  = hi ? 8 : 0;               // B-matrix K base (ISA layout)

    const unsigned short* qrow = qb + (size_t)(b * LL + t0 + m) * DD;

    #pragma unroll 1
    for (int pass = 0; pass < 2; ++pass) {
        const int dbase = pass * 256;          // 8 chunks of 32 per pass

        // A half-tile (16 x 256 bf16) held in registers, reused for all s.
        ABTile a[8];
        #pragma unroll
        for (int kc = 0; kc < 8; ++kc) {
            a[kc].q[0] = *(const uint4*)(qrow + dbase + kc * 32 + ka);
            a[kc].q[1] = *(const uint4*)(qrow + dbase + kc * 32 + ka + 16);
        }

        #pragma unroll 1
        for (int st = wave; st < 256; st += 8) {   // uniform per wave
            const int s0 = st << 4;
            const unsigned short* krow =
                kb + (size_t)(b * LL + s0 + m) * DD + dbase;

            if (st + 8 < 256) {                    // uniform branch
                // next s-tile for this wave: 128 rows ahead
                __builtin_prefetch(krow + (size_t)128 * DD, 0, 0);
            }

            v8f c = {};
            #pragma unroll
            for (int g = 0; g < 2; ++g) {
                // stage 4 B-chunks (8 x b128 loads in flight), then 4 WMMAs
                ABTile bt[4];
                #pragma unroll
                for (int j = 0; j < 4; ++j) {
                    const int d0 = (g * 4 + j) * 32;
                    bt[j].q[0] = *(const uint4*)(krow + d0 + kbb);
                    bt[j].q[1] = *(const uint4*)(krow + d0 + kbb + 16);
                }
                #pragma unroll
                for (int j = 0; j < 4; ++j) {
                    c = __builtin_amdgcn_wmma_f32_16x16x32_bf16(
                            false, a[g * 4 + j].v, false, bt[j].v,
                            (short)0, c, false, false);
                }
            }

            // C/D layout: VGPR r -> M = r + 8*hi, N = lane&15.
            // G[t0+M, s0+N] contributes to lag (t - s) mod L.
            #pragma unroll
            for (int r = 0; r < 8; ++r) {
                const int M   = r + (hi << 3);
                const int lag = (t0 + M - s0 - m) & LMASK;
                atomicAdd(&lagAcc[lag], c[r]);
            }
        }
    }
    __syncthreads();

    const float scale = 1.0f / (float)DD;      // mean over (h,e)
    for (int i = tid; i < LL; i += 256)
        atomicAdd(&meanval[b * LL + i], lagAcc[i] * scale);
}

// ---------------------------------------------------------------------------
// Kernel 3: batch-mean over lags, top-8 lags, per-batch softmax weights.
// single block of 256 threads.
// ---------------------------------------------------------------------------
__global__ void ac_topk_softmax(const float* __restrict__ meanval,
                                int* __restrict__ idxOut,
                                float* __restrict__ wOut) {
    __shared__ float S[LL];
    __shared__ float rv[256];
    __shared__ int   ri[256];
    __shared__ int   sel[TOPK];

    const int t = threadIdx.x;
    for (int l = t; l < LL; l += 256) {
        float s = 0.0f;
        #pragma unroll
        for (int b = 0; b < BB; ++b) s += meanval[b * LL + l];
        S[l] = s;
    }
    __syncthreads();

    for (int k = 0; k < TOPK; ++k) {
        float best = -3.4e38f;
        int   bi   = 0;
        for (int l = t; l < LL; l += 256)
            if (S[l] > best) { best = S[l]; bi = l; }
        rv[t] = best; ri[t] = bi;
        __syncthreads();
        for (int off = 128; off > 0; off >>= 1) {
            if (t < off && rv[t + off] > rv[t]) { rv[t] = rv[t + off]; ri[t] = ri[t + off]; }
            __syncthreads();
        }
        if (t == 0) { sel[k] = ri[0]; S[ri[0]] = -3.4e38f; }
        __syncthreads();
    }

    if (t < TOPK) idxOut[t] = sel[t];
    if (t < BB) {
        float w[TOPK];
        float mx = -3.4e38f;
        #pragma unroll
        for (int k = 0; k < TOPK; ++k) {
            w[k] = meanval[t * LL + sel[k]];
            mx = fmaxf(mx, w[k]);
        }
        float sum = 0.0f;
        #pragma unroll
        for (int k = 0; k < TOPK; ++k) { w[k] = __expf(w[k] - mx); sum += w[k]; }
        const float inv = 1.0f / sum;
        #pragma unroll
        for (int k = 0; k < TOPK; ++k) wOut[t * TOPK + k] = w[k] * inv;
    }
}

// ---------------------------------------------------------------------------
// Kernel 4: out[b,l,:] = sum_k w[b,k] * V[b,(l+idx[k])%L,:]   (float4 gather)
// grid: B*L*(D/4) threads.
// ---------------------------------------------------------------------------
__global__ void ac_agg(const float* __restrict__ v,
                       const int* __restrict__ idx,
                       const float* __restrict__ w,
                       float* __restrict__ out) {
    const unsigned int tid = blockIdx.x * 256u + threadIdx.x;
    const int d4 = tid & 127;                 // D/4 = 128 float4 per row
    const int l  = (tid >> 7) & LMASK;
    const int b  = tid >> 19;

    int   sh[TOPK];
    float ww[TOPK];
    #pragma unroll
    for (int k = 0; k < TOPK; ++k) { sh[k] = idx[k]; ww[k] = w[b * TOPK + k]; }

    float4 acc = make_float4(0.f, 0.f, 0.f, 0.f);
    #pragma unroll
    for (int k = 0; k < TOPK; ++k) {
        const int pos = (l + sh[k]) & LMASK;
        const float4 vv = *(const float4*)(v + ((size_t)(b * LL + pos) * DD) + d4 * 4);
        acc.x += ww[k] * vv.x;
        acc.y += ww[k] * vv.y;
        acc.z += ww[k] * vv.z;
        acc.w += ww[k] * vv.w;
    }
    *(float4*)(out + ((size_t)(b * LL + l) * DD) + d4 * 4) = acc;
}

// ---------------------------------------------------------------------------
// Host-side launcher (graph-capture safe: only kernel launches on `stream`).
// Workspace layout (bytes):
//   [0, 128K)          mean_value  float[B*L]
//   [128K, +32M)       qb  bf16[B*L*D]
//   [.., +32M)         kb  bf16[B*L*D]
//   then idx int[8], weights float[B*8]
// ---------------------------------------------------------------------------
extern "C" void kernel_launch(void* const* d_in, const int* in_sizes, int n_in,
                              void* d_out, int out_size, void* d_ws, size_t ws_size,
                              hipStream_t stream) {
    (void)in_sizes; (void)n_in; (void)out_size; (void)ws_size;

    const float* q = (const float*)d_in[0];
    const float* k = (const float*)d_in[1];
    const float* v = (const float*)d_in[2];
    float* out = (float*)d_out;

    char* ws = (char*)d_ws;
    const size_t packBytes = (size_t)BB * LL * DD * 2;        // 32 MiB each
    float*          meanval = (float*)ws;
    unsigned short* qb      = (unsigned short*)(ws + 131072);
    unsigned short* kb      = (unsigned short*)(ws + 131072 + packBytes);
    int*            idx     = (int*)(ws + 131072 + 2 * packBytes);
    float*          wgt     = (float*)(ws + 131072 + 2 * packBytes + 32);

    const int nConv = (BB * LL * DD) / 4;                      // 4,194,304
    ac_prep<<<nConv / 256, 256, 0, stream>>>(q, k, qb, kb, meanval);

    ac_corr_wmma<<<BB * (LL / 16), 256, 0, stream>>>(qb, kb, meanval);

    ac_topk_softmax<<<1, 256, 0, stream>>>(meanval, idx, wgt);

    const int nAgg = BB * LL * (DD / 4);                       // 4,194,304
    ac_agg<<<nAgg / 256, 256, 0, stream>>>(v, idx, wgt, out);
}
